// QuantLinear_17729624998040
// MI455X (gfx1250) — matrix operand order repaired and checked by reading, not compiled, not run on hardware
//
#include <hip/hip_runtime.h>
#include <stdint.h>

// ---------------------------------------------------------------------------
// Fake-quant linear for MI455X (gfx1250, wave32):
//   out = fq_out( x @ fq_w(W)^T + bias ),  M=8192, N=4096, K=4096 (ref shape)
//
// Fast path (needs ~101 MB workspace):
//   1) prepass: x (f32) -> f16 in ws          (memory bound)
//   2) prepass: fq_w(W) (f32) -> f16 in ws    (memory bound; exact in f16)
//   3) GEMM: async global->LDS B128 copies (ASYNCcnt DMA, no VALU staging)
//      feeding v_wmma_f32_16x16x32_f16, 128x128 tile, 8 waves, double buffer.
// Fallback path (small ws): fused kernel that quantizes during staging.
// ---------------------------------------------------------------------------

typedef __attribute__((ext_vector_type(16))) _Float16 v16h;
typedef __attribute__((ext_vector_type(8)))  _Float16 v8h;
typedef __attribute__((ext_vector_type(8)))  float    v8f;
typedef __attribute__((ext_vector_type(4)))  float    v4f;
typedef __attribute__((ext_vector_type(4)))  int      v4i;

// pointer types matching the async-to-LDS builtin prototype
typedef __attribute__((address_space(1))) v4i* gv4i_p;   // global (device) v4i*
typedef __attribute__((address_space(3))) v4i* lv4i_p;   // LDS v4i*

#define BM  128
#define BN  128
#define BK  32
#define LDP (BK + 8)   // LDS pitch in halves (80 B rows, 16B-aligned chunks)

#if __has_builtin(__builtin_amdgcn_global_load_async_to_lds_b128)
#define USE_ASYNC_LDS 1
#else
#define USE_ASYNC_LDS 0
#endif

static __device__ __forceinline__ void wait_async0() {
#if __has_builtin(__builtin_amdgcn_s_wait_asynccnt)
    __builtin_amdgcn_s_wait_asynccnt(0);
#else
    asm volatile("s_wait_asynccnt 0x0" ::: "memory");
#endif
}

static __device__ __forceinline__ float fq1(float v, float rs, float s, float z) {
    float q = __builtin_rintf(v * rs + z);
    q = fminf(fmaxf(q, -128.0f), 127.0f);
    return (q - z) * s;
}

// ------------------------- prepass: x f32 -> f16 ---------------------------
__global__ __launch_bounds__(256)
void xcvt_f16(const float* __restrict__ x, _Float16* __restrict__ xh, size_t n8) {
    const size_t i = ((size_t)blockIdx.x * blockDim.x + threadIdx.x);
    if (i >= n8) return;
    const size_t b = i * 8;
    v4f a = *(const v4f*)(x + b);
    v4f c = *(const v4f*)(x + b + 4);
    v8h h;
#pragma unroll
    for (int j = 0; j < 4; ++j) { h[j] = (_Float16)a[j]; h[j + 4] = (_Float16)c[j]; }
    *(v8h*)(xh + b) = h;
}

// ---------------- prepass: fake-quant W f32 -> f16 (per-row) ---------------
__global__ __launch_bounds__(256)
void wquant_f16(const float* __restrict__ w, const float* __restrict__ w_scale,
                const float* __restrict__ w_zero, _Float16* __restrict__ wh, int K) {
    const int n = blockIdx.y;
    const int base = (blockIdx.x * 256 + threadIdx.x) * 8;
    if (base >= K) return;
    const float s = w_scale[n], z = w_zero[n], rs = 1.0f / s;
    const float* src = w + (size_t)n * K + base;
    v4f a = *(const v4f*)src;
    v4f c = *(const v4f*)(src + 4);
    v8h h;
#pragma unroll
    for (int j = 0; j < 4; ++j) {
        h[j]     = (_Float16)fq1(a[j], rs, s, z);
        h[j + 4] = (_Float16)fq1(c[j], rs, s, z);
    }
    *(v8h*)(wh + (size_t)n * K + base) = h;
}

// ------------------------- GEMM on pre-converted f16 -----------------------
__global__ __launch_bounds__(256)
void qgemm_f16(const _Float16* __restrict__ A,   // [M,K] f16
               const _Float16* __restrict__ B,   // [N,K] f16 (fake-quantized)
               const float* __restrict__ bias,
               const float* __restrict__ out_scale,
               const float* __restrict__ out_zero,
               float* __restrict__ out, int M, int N, int K)
{
    __shared__ _Float16 As[2][BM * LDP];
    __shared__ _Float16 Bs[2][BN * LDP];

    const int tid  = threadIdx.x;
    const int lane = tid & 31;
    const int wid  = tid >> 5;
    const int wm   = wid >> 2;          // 0..1
    const int wn   = wid & 3;           // 0..3
    const int bn0  = blockIdx.x * BN;
    const int bm0  = blockIdx.y * BM;

    // staging map: rows of 64B (32 halves); 4 x 16B chunks per row
    const int c = tid & 3;              // chunk -> halves 8*c
    const int r = tid >> 2;             // row 0..63 (+64)

    auto stage = [&](int kt, int buf) {
#pragma unroll
        for (int i = 0; i < 2; ++i) {
            const int row = r + 64 * i;
            const _Float16* gA = A + (size_t)(bm0 + row) * K + kt * BK + c * 8;
            const _Float16* gB = B + (size_t)(bn0 + row) * K + kt * BK + c * 8;
            _Float16* lA = &As[buf][row * LDP + c * 8];
            _Float16* lB = &Bs[buf][row * LDP + c * 8];
#if USE_ASYNC_LDS
            __builtin_amdgcn_global_load_async_to_lds_b128(
                (gv4i_p)gA, (lv4i_p)lA, 0, 0);
            __builtin_amdgcn_global_load_async_to_lds_b128(
                (gv4i_p)gB, (lv4i_p)lB, 0, 0);
#else
            *(v8h*)lA = *(const v8h*)gA;
            *(v8h*)lB = *(const v8h*)gB;
#endif
        }
    };

    v8f acc[4][2];
#pragma unroll
    for (int tm = 0; tm < 4; ++tm)
#pragma unroll
        for (int tn = 0; tn < 2; ++tn)
            acc[tm][tn] = (v8f)(0.0f);

    const int lr = lane & 15;
    const int hi = lane >> 4;

    auto compute = [&](int buf) {
        v16h af[4], bf[2];
#pragma unroll
        for (int tm = 0; tm < 4; ++tm) {
            const _Float16* ap = &As[buf][(wm * 64 + tm * 16 + lr) * LDP + hi * 8];
            v8h a0 = *(const v8h*)ap;
            v8h a1 = *(const v8h*)(ap + 16);
            af[tm] = __builtin_shufflevector(a0, a1,
                0, 1, 2, 3, 4, 5, 6, 7, 8, 9, 10, 11, 12, 13, 14, 15);
        }
#pragma unroll
        for (int tn = 0; tn < 2; ++tn) {
            const _Float16* bp = &Bs[buf][(wn * 32 + tn * 16 + lr) * LDP + hi * 8];
            v8h b0 = *(const v8h*)bp;
            v8h b1 = *(const v8h*)(bp + 16);
            bf[tn] = __builtin_shufflevector(b0, b1,
                0, 1, 2, 3, 4, 5, 6, 7, 8, 9, 10, 11, 12, 13, 14, 15);
        }
#pragma unroll
        for (int tm = 0; tm < 4; ++tm)
#pragma unroll
            for (int tn = 0; tn < 2; ++tn)
                acc[tm][tn] = __builtin_amdgcn_wmma_f32_16x16x32_f16(
                    false, af[tm], false, bf[tn],
                    (short)0, acc[tm][tn], false, false);
    };

    const int NK = K / BK;
    stage(0, 0);
    for (int kt = 0; kt < NK; ++kt) {
        wait_async0();
        __syncthreads();
        if (kt + 1 < NK) stage(kt + 1, (kt + 1) & 1);  // DMA overlaps WMMA
        compute(kt & 1);
    }

    const float os  = out_scale[0];
    const float oz  = out_zero[0];
    const float ors = 1.0f / os;
#pragma unroll
    for (int tm = 0; tm < 4; ++tm) {
#pragma unroll
        for (int tn = 0; tn < 2; ++tn) {
            const int n = bn0 + wn * 32 + tn * 16 + lr;
            const float bv = bias[n];
#pragma unroll
            for (int rr = 0; rr < 8; ++rr) {
                const int m = bm0 + wm * 64 + tm * 16 + hi * 8 + rr;
                float v = acc[tm][tn][rr] + bv;
                out[(size_t)m * N + n] = fq1(v, ors, os, oz);
            }
        }
    }
}

// --------------------- fallback: fused (small workspace) -------------------
__global__ __launch_bounds__(256)
void qlinear_fused(const float* __restrict__ x, const float* __restrict__ w,
                   const float* __restrict__ bias, const float* __restrict__ w_scale,
                   const float* __restrict__ w_zero, const float* __restrict__ out_scale,
                   const float* __restrict__ out_zero, float* __restrict__ out,
                   int M, int N, int K)
{
    __shared__ _Float16 As[2][BM * LDP];
    __shared__ _Float16 Bs[2][BN * LDP];

    const int tid  = threadIdx.x;
    const int lane = tid & 31;
    const int wid  = tid >> 5;
    const int wm   = wid >> 2;
    const int wn   = wid & 3;
    const int bn0  = blockIdx.x * BN;
    const int bm0  = blockIdx.y * BM;
    const int cg   = tid & 7;
    const int r0   = tid >> 3;

    float sc[4], zp[4], rs[4];
#pragma unroll
    for (int i = 0; i < 4; ++i) {
        const int n = bn0 + r0 + 32 * i;
        sc[i] = w_scale[n]; zp[i] = w_zero[n]; rs[i] = 1.0f / sc[i];
    }

    v8f acc[4][2];
#pragma unroll
    for (int tm = 0; tm < 4; ++tm)
#pragma unroll
        for (int tn = 0; tn < 2; ++tn) acc[tm][tn] = (v8f)(0.0f);

    const int NK = K / BK;
    v4f ra[4], rb[4];

    auto load_tiles = [&](int kt) {
        const float* xa = x + (size_t)bm0 * K + (size_t)kt * BK;
        const float* wb = w + (size_t)bn0 * K + (size_t)kt * BK;
#pragma unroll
        for (int i = 0; i < 4; ++i) {
            const int row = r0 + 32 * i;
            ra[i] = *(const v4f*)(xa + (size_t)row * K + 4 * cg);
            rb[i] = *(const v4f*)(wb + (size_t)row * K + 4 * cg);
        }
    };
    auto store_tiles = [&](int buf) {
#pragma unroll
        for (int i = 0; i < 4; ++i) {
            const int row = r0 + 32 * i;
            _Float16* ap = &As[buf][row * LDP + 4 * cg];
            _Float16* bp = &Bs[buf][row * LDP + 4 * cg];
#pragma unroll
            for (int j = 0; j < 4; ++j) {
                ap[j] = (_Float16)ra[i][j];
                bp[j] = (_Float16)fq1(rb[i][j], rs[i], sc[i], zp[i]);
            }
        }
    };

    const int lr = lane & 15;
    const int hi = lane >> 4;
    auto compute = [&](int buf) {
        v16h af[4], bf[2];
#pragma unroll
        for (int tm = 0; tm < 4; ++tm) {
            const _Float16* ap = &As[buf][(wm * 64 + tm * 16 + lr) * LDP + hi * 8];
            v8h a0 = *(const v8h*)ap;
            v8h a1 = *(const v8h*)(ap + 16);
            af[tm] = __builtin_shufflevector(a0, a1,
                0, 1, 2, 3, 4, 5, 6, 7, 8, 9, 10, 11, 12, 13, 14, 15);
        }
#pragma unroll
        for (int tn = 0; tn < 2; ++tn) {
            const _Float16* bp = &Bs[buf][(wn * 32 + tn * 16 + lr) * LDP + hi * 8];
            v8h b0 = *(const v8h*)bp;
            v8h b1 = *(const v8h*)(bp + 16);
            bf[tn] = __builtin_shufflevector(b0, b1,
                0, 1, 2, 3, 4, 5, 6, 7, 8, 9, 10, 11, 12, 13, 14, 15);
        }
#pragma unroll
        for (int tm = 0; tm < 4; ++tm)
#pragma unroll
            for (int tn = 0; tn < 2; ++tn)
                acc[tm][tn] = __builtin_amdgcn_wmma_f32_16x16x32_f16(
                    false, af[tm], false, bf[tn], (short)0, acc[tm][tn], false, false);
    };

    load_tiles(0);
    store_tiles(0);
    for (int kt = 0; kt < NK; ++kt) {
        __syncthreads();
        if (kt + 1 < NK) load_tiles(kt + 1);
        compute(kt & 1);
        if (kt + 1 < NK) store_tiles((kt + 1) & 1);
    }

    const float os = out_scale[0], oz = out_zero[0], ors = 1.0f / os;
#pragma unroll
    for (int tm = 0; tm < 4; ++tm)
#pragma unroll
        for (int tn = 0; tn < 2; ++tn) {
            const int n = bn0 + wn * 32 + tn * 16 + lr;
            const float bv = bias[n];
#pragma unroll
            for (int rr = 0; rr < 8; ++rr) {
                const int m = bm0 + wm * 64 + tm * 16 + hi * 8 + rr;
                out[(size_t)m * N + n] = fq1(acc[tm][tn][rr] + bv, ors, os, oz);
            }
        }
}

extern "C" void kernel_launch(void* const* d_in, const int* in_sizes, int n_in,
                              void* d_out, int out_size, void* d_ws, size_t ws_size,
                              hipStream_t stream) {
    const float* x         = (const float*)d_in[0];
    const float* w         = (const float*)d_in[1];
    const float* bias      = (const float*)d_in[2];
    const float* w_scale   = (const float*)d_in[3];
    const float* w_zero    = (const float*)d_in[4];
    const float* out_scale = (const float*)d_in[5];
    const float* out_zero  = (const float*)d_in[6];

    const int N = in_sizes[2];
    const int K = in_sizes[1] / N;
    const int M = in_sizes[0] / K;

    const size_t needA = (size_t)M * K * sizeof(_Float16);
    const size_t needB = (size_t)N * K * sizeof(_Float16);
    dim3 grid(N / BN, M / BM);

    if (ws_size >= needA + needB) {
        _Float16* Ah = (_Float16*)d_ws;
        _Float16* Bh = (_Float16*)((char*)d_ws + needA);

        const size_t n8 = (size_t)M * K / 8;
        xcvt_f16<<<(unsigned)((n8 + 255) / 256), 256, 0, stream>>>(x, Ah, n8);

        dim3 wg((K + 2047) / 2048, N);
        wquant_f16<<<wg, 256, 0, stream>>>(w, w_scale, w_zero, Bh, K);

        qgemm_f16<<<grid, 256, 0, stream>>>(Ah, Bh, bias, out_scale, out_zero,
                                            (float*)d_out, M, N, K);
    } else {
        qlinear_fused<<<grid, 256, 0, stream>>>(x, w, bias, w_scale, w_zero,
                                                out_scale, out_zero,
                                                (float*)d_out, M, N, K);
    }
}
